// GraphUNetEncoder_89996744720773
// MI455X (gfx1250) — matrix operand order repaired and checked by reading, not compiled, not run on hardware
//
#include <hip/hip_runtime.h>
#include <hip/hip_bf16.h>
#include <math.h>

#define NNODES 4096
#define CH     256
#define NCLS   10

typedef unsigned int u32;
typedef __attribute__((ext_vector_type(2)))  float  v2f;
typedef __attribute__((ext_vector_type(8)))  float  v8f;
typedef __attribute__((ext_vector_type(16))) __bf16 v16bf;

union FragB16 { v16bf v; u32 u[8]; uint4 q[2]; };
union PairBF  { __bf16 h[2]; u32 u; };

// ---------------------------------------------------------------- utilities

__global__ void k_zero(float* p, size_t n) {
  size_t i = (size_t)blockIdx.x * blockDim.x + threadIdx.x;
  if (i < n) p[i] = 0.0f;
}

__global__ void k_scatter_edges(const int* ei, int E, float* A, int n) {
  int e = blockIdx.x * blockDim.x + threadIdx.x;
  if (e < E) {
    int s = ei[e];
    int d = ei[E + e];
    atomicAdd(&A[(size_t)s * n + d], 1.0f);   // integer sums -> exact & deterministic
  }
}

__global__ void k_diag_fix(float* A, int n, int ld, float fill) {
  int i = blockIdx.x * blockDim.x + threadIdx.x;
  if (i < n) {
    size_t o = (size_t)i * ld + i;
    if (A[o] == 0.0f) A[o] = fill;
  }
}

__global__ void k_diag_set(float* A, int n, int ld, float val) {
  int i = blockIdx.x * blockDim.x + threadIdx.x;
  if (i < n) A[(size_t)i * ld + i] = val;
}

// colsum of (diag-fixed) A -> dinv = rsqrt(deg) (0 where deg<=0)
__global__ void k_colsum_rsqrt(const float* A, int n, int ld, float* dinv) {
  int c = blockIdx.x * blockDim.x + threadIdx.x;
  if (c >= n) return;
  float s = 0.0f;
  for (int j = 0; j < n; ++j) s += A[(size_t)j * ld + c];
  dinv[c] = (s > 0.0f) ? rsqrtf(s) : 0.0f;
}

// f32 A -> (a) row-major bf16 copy (A-fragment side) and
//          (b) K-pair-interleaved packed form (B-fragment side),
// with the diagonal forced to 1.0 (augment's set-diag).  Adjacency entries are
// small integers -> exact in bf16.  One thread per (row-pair, column).
__global__ void k_pack_bf16_diag1(const float* A, __bf16* Abf, u32* Apk, int n) {
  size_t i = (size_t)blockIdx.x * blockDim.x + threadIdx.x;
  size_t tot = (size_t)(n / 2) * n;
  if (i >= tot) return;
  int kp = (int)(i / (size_t)n);
  int c  = (int)(i % (size_t)n);
  int r0 = 2 * kp, r1 = 2 * kp + 1;
  float v0 = (r0 == c) ? 1.0f : A[(size_t)r0 * n + c];
  float v1 = (r1 == c) ? 1.0f : A[(size_t)r1 * n + c];
  PairBF p;
  p.h[0] = (__bf16)v0;
  p.h[1] = (__bf16)v1;
  Abf[(size_t)r0 * n + c] = p.h[0];
  Abf[(size_t)r1 * n + c] = p.h[1];
  Apk[(size_t)kp * n + c] = p.u;
}

__global__ void k_row_scale(const float* Y, const float* dinv, float* Z, int n, int c) {
  int i = blockIdx.x * blockDim.x + threadIdx.x;
  if (i < n * c) Z[i] = Y[i] * dinv[i / c];
}

__global__ void k_bias_relu_scale(const float* G, const float* dinv, const float* bias,
                                  float* X, int n, int c) {
  int i = blockIdx.x * blockDim.x + threadIdx.x;
  if (i < n * c) {
    float v = G[i] * dinv[i / c] + bias[i % c];
    X[i] = fmaxf(v, 0.0f);
  }
}

// ---------------------------------------------------------------- pooling

__global__ void k_vec_norm(const float* p, float* out) {   // block of 256, C==256
  __shared__ float red[256];
  int t = threadIdx.x;
  float v = p[t];
  red[t] = v * v;
  __syncthreads();
  for (int st = 128; st > 0; st >>= 1) {
    if (t < st) red[t] += red[t + st];
    __syncthreads();
  }
  if (t == 0) out[0] = sqrtf(red[0]);
}

__global__ void k_row_score(const float* X, const float* p, const float* pnorm,
                            float* s, int c) {              // one block (256) per row
  __shared__ float red[256];
  int t = threadIdx.x;
  int row = blockIdx.x;
  red[t] = X[(size_t)row * c + t] * p[t];
  __syncthreads();
  for (int st = 128; st > 0; st >>= 1) {
    if (t < st) red[t] += red[t + st];
    __syncthreads();
  }
  if (t == 0) s[row] = tanhf(red[0] / pnorm[0]);
}

// stable descending rank == jax.lax.top_k ordering (ties: smaller index first)
__global__ void k_topk_rank(const float* s, int n, int k, int* perm, float* vals) {
  int i = blockIdx.x * blockDim.x + threadIdx.x;
  if (i >= n) return;
  float si = s[i];
  int r = 0;
  for (int j = 0; j < n; ++j) {
    float sj = s[j];
    r += (sj > si) || (sj == si && j < i);
  }
  if (r < k) { perm[r] = i; vals[r] = si; }
}

__global__ void k_gather_x(const float* X, const int* perm, const float* vals,
                           float* Xp, int c) {              // grid k, block 256 (==C)
  int r = blockIdx.x;
  int t = threadIdx.x;
  Xp[(size_t)r * c + t] = X[(size_t)perm[r] * c + t] * vals[r];
}

__global__ void k_gather_adj(const float* B2, const int* perm, int ldb,
                             float* Ap, int k) {            // grid (k/256, k)
  int r = blockIdx.y;
  int s = blockIdx.x * blockDim.x + threadIdx.x;
  if (s < k) Ap[(size_t)r * k + s] = B2[(size_t)perm[r] * ldb + perm[s]];
}

// ---------------------------------------------------------------- bf16 WMMA GEMM
// D(f32) = A(bf16) * B(bf16), B supplied pre-packed as K-pair dwords.
// Block = 128 threads (4 waves), block tile 32x32, K chunks of 32 staged through
// LDS with b128 loads/stores only; one v_wmma_f32_16x16x32_bf16 per wave per chunk.
// All dims multiples of 32.
__global__ void gemm_bf16_wmma(const __bf16* __restrict__ Ag,   // row-major bf16
                               const u32*    __restrict__ Bpk,  // [K/2][N] packed pairs
                               float*        __restrict__ Dg,
                               int Kdim, int lda, int ldbp, int ldd) {
  __shared__ u32 As2[32][20];   // [m][kpair], row stride 80B (16B multiple)
  __shared__ u32 Bs2[16][36];   // [kpair][n], row stride 144B (16B multiple)
  const int m0   = blockIdx.y * 32;
  const int n0   = blockIdx.x * 32;
  const int tid  = threadIdx.x;
  const int lane = tid & 31;
  const int wav  = tid >> 5;
  const int wm   = (wav & 1) * 16;
  const int wn   = (wav >> 1) * 16;
  const int half = lane >> 4;
  const int lr   = lane & 15;

  const int am = tid >> 2, ag = (tid & 3) * 4;   // A fill: 32 rows x 4 uint4
  const int br = tid >> 3, bg = (tid & 7) * 4;   // B fill: 16 rows x 8 uint4

  v8f acc = {0.f, 0.f, 0.f, 0.f, 0.f, 0.f, 0.f, 0.f};

  for (int kk = 0; kk < Kdim; kk += 32) {
    // A panel: 32x32 bf16 = 32x16 dwords; one b128 per thread
    const u32* asrc = (const u32*)Ag + (((size_t)(m0 + am) * lda + kk) >> 1);
    *(uint4*)&As2[am][ag] = *(const uint4*)(asrc + ag);
    // B panel (pre-packed): 16x32 dwords; one b128 per thread
    const u32* bsrc = Bpk + (size_t)((kk >> 1) + br) * ldbp + n0;
    *(uint4*)&Bs2[br][bg] = *(const uint4*)(bsrc + bg);

    if (kk + 32 < Kdim) {                        // global_prefetch_b8 next panels
      __builtin_prefetch((const u32*)Ag + (((size_t)(m0 + am) * lda + kk + 32) >> 1), 0, 1);
      __builtin_prefetch(Bpk + (size_t)(((kk + 32) >> 1) + br) * ldbp + n0, 0, 1);
    }
    __syncthreads();

    FragB16 fa, fb;
    // A frag: K = kk + half*8 + {0..7} and kk + 16 + half*8 + {0..7} -> 2x b128
    fa.q[0] = *(const uint4*)&As2[wm + lr][half * 4];
    fa.q[1] = *(const uint4*)&As2[wm + lr][8 + half * 4];
    // B frag: K = kk + half*8 + {0..7} and kk + 16 + half*8 + {0..7} -> 8x b32
#pragma unroll
    for (int j = 0; j < 8; ++j) fb.u[j] = Bs2[(j >> 2) * 8 + half * 4 + (j & 3)][wn + lr];

    acc = __builtin_amdgcn_wmma_f32_16x16x32_bf16(false, fa.v, false, fb.v,
                                                  (short)0, acc, false, false);
    __syncthreads();
  }

#pragma unroll
  for (int v = 0; v < 8; ++v)                    // C/D: M = v + 8*half, N = lr
    Dg[(size_t)(m0 + wm + v + half * 8) * ldd + (n0 + wn + lr)] = acc[v];
}

// ---------------------------------------------------------------- f32 WMMA GEMM
// D = op(A) * B with op(A)[m,k] = TRANSA ? A[k*lda+m] : A[m*lda+k].
// V_WMMA_F32_16X16X4_F32 (full f32 precision) for the GCN GEMMs (minor FLOPs).
template <int TRANSA>
__global__ void gemm_f32_wmma(const float* __restrict__ Ag,
                              const float* __restrict__ Bg,
                              float* __restrict__ Dg,
                              int Kdim, int lda, int ldb, int ldd) {
  __shared__ float As[32][36];   // row stride 144B (16B multiple)
  __shared__ float Bs[32][36];
  const int m0   = blockIdx.y * 32;
  const int n0   = blockIdx.x * 32;
  const int tid  = threadIdx.x;
  const int lane = tid & 31;
  const int wav  = tid >> 5;
  const int wm   = (wav & 1) * 16;
  const int wn   = (wav >> 1) * 16;
  const int half = lane >> 4;
  const int lr   = lane & 15;

  v8f acc = {0.f, 0.f, 0.f, 0.f, 0.f, 0.f, 0.f, 0.f};

  for (int kk = 0; kk < Kdim; kk += 32) {
#pragma unroll
    for (int it = 0; it < 2; ++it) {             // 32x32 floats = 256 float4
      int w = tid + it * 128;
      if (TRANSA) {                              // read columns of A, b128 along m
        int k = w & 31, mg = (w >> 5) * 4;
        float4 t4 = *(const float4*)&Ag[(size_t)(kk + k) * lda + (m0 + mg)];
        As[mg + 0][k] = t4.x; As[mg + 1][k] = t4.y;
        As[mg + 2][k] = t4.z; As[mg + 3][k] = t4.w;
      } else {
        int m = w >> 3, g = (w & 7) * 4;
        *(float4*)&As[m][g] = *(const float4*)&Ag[(size_t)(m0 + m) * lda + (kk + g)];
      }
      int bk = w >> 3, bgc = (w & 7) * 4;
      *(float4*)&Bs[bk][bgc] = *(const float4*)&Bg[(size_t)(kk + bk) * ldb + (n0 + bgc)];
    }
    __syncthreads();

#pragma unroll
    for (int t = 0; t < 8; ++t) {                // 8 x (16x16x4) per 32-K chunk
      int kb = t * 4 + half * 2;                 // 32-bit A/B frag: K = 2*half + v
      v2f a = *(const v2f*)&As[wm + lr][kb];     // b64 (8B aligned)
      v2f b;
      b.x = Bs[kb][wn + lr];
      b.y = Bs[kb + 1][wn + lr];
      acc = __builtin_amdgcn_wmma_f32_16x16x4_f32(false, a, false, b,
                                                  (short)0, acc, false, false);
    }
    __syncthreads();
  }

#pragma unroll
  for (int v = 0; v < 8; ++v)
    Dg[(size_t)(m0 + wm + v + half * 8) * ldd + (n0 + wn + lr)] = acc[v];
}

// ---------------------------------------------------------------- readout

__global__ void k_col_mean(const float* X, float* emb, int n, int c) { // grid C
  __shared__ float red[256];
  int t  = threadIdx.x;
  int ch = blockIdx.x;
  float acc = 0.0f;
  for (int i = t; i < n; i += 256) acc += X[(size_t)i * c + ch];
  red[t] = acc;
  __syncthreads();
  for (int st = 128; st > 0; st >>= 1) {
    if (t < st) red[t] += red[t + st];
    __syncthreads();
  }
  if (t == 0) emb[ch] = red[0] / (float)n;
}

__global__ void k_finalize(const float* emb, const float* wc, const float* bc,
                           float* out, int c, int ncls) {   // single block of 256
  __shared__ float red[256];
  __shared__ float en[256];
  __shared__ float lg[16];
  int t = threadIdx.x;
  float e = emb[t];
  red[t] = e * e;
  __syncthreads();
  for (int st = 128; st > 0; st >>= 1) {
    if (t < st) red[t] += red[t + st];
    __syncthreads();
  }
  float scale = 1.0f / fmaxf(sqrtf(red[0]), 1e-12f);
  float env = e * scale;
  en[t] = env;
  out[t] = env;                                    // normalized embedding (1 x C)
  __syncthreads();
  if (t < ncls) {
    float acc = bc[t];
    for (int q = 0; q < c; ++q) acc += en[q] * wc[t * c + q];
    lg[t] = acc;
  }
  __syncthreads();
  if (t == 0) {
    float mx = -1e30f;
    for (int q = 0; q < ncls; ++q) mx = fmaxf(mx, lg[q]);
    float se = 0.0f;
    for (int q = 0; q < ncls; ++q) se += expf(lg[q] - mx);
    float lse = logf(se);
    for (int q = 0; q < ncls; ++q) out[c + q] = lg[q] - mx - lse;
  }
}

// ---------------------------------------------------------------- host side

static void launch_gcn(hipStream_t stream, float* A, int n,
                       const float* Xin, const float* W, const float* bias,
                       float* Xout, float* Y, float* Z, float* dinv) {
  const int T = 256;
  k_diag_fix<<<(n + T - 1) / T, T, 0, stream>>>(A, n, n, 2.0f);
  k_colsum_rsqrt<<<(n + T - 1) / T, T, 0, stream>>>(A, n, n, dinv);
  // Y = Xin @ W   (n x CH x CH)
  gemm_f32_wmma<0><<<dim3(CH / 32, n / 32), 128, 0, stream>>>(Xin, W, Y, CH, CH, CH, CH);
  // Z = dinv[i] * Y
  k_row_scale<<<(n * CH + T - 1) / T, T, 0, stream>>>(Y, dinv, Z, n, CH);
  // G = A^T @ Z   (n x CH x n)  -> into Y
  gemm_f32_wmma<1><<<dim3(CH / 32, n / 32), 128, 0, stream>>>(A, Z, Y, n, n, CH, CH);
  // Xout = relu(dinv[i] * G + b)
  k_bias_relu_scale<<<(n * CH + T - 1) / T, T, 0, stream>>>(Y, dinv, bias, Xout, n, CH);
}

extern "C" void kernel_launch(void* const* d_in, const int* in_sizes, int n_in,
                              void* d_out, int out_size, void* d_ws, size_t ws_size,
                              hipStream_t stream) {
  const float* xin = (const float*)d_in[0];
  const int*   ei  = (const int*)d_in[1];
  const float* w0  = (const float*)d_in[2];
  const float* b0  = (const float*)d_in[3];
  const float* w1  = (const float*)d_in[4];
  const float* b1  = (const float*)d_in[5];
  const float* w2  = (const float*)d_in[6];
  const float* b2  = (const float*)d_in[7];
  const float* p1  = (const float*)d_in[8];
  const float* p2  = (const float*)d_in[9];
  const float* wc  = (const float*)d_in[10];
  const float* bc  = (const float*)d_in[11];
  const int E = in_sizes[1] / 2;
  const int N = NNODES;
  const size_t NN = (size_t)N * N;

  // workspace layout
  float*  Abuf  = (float*)d_ws;              // adjacency (f32), stride = current n
  float*  Bbuf  = Abuf + NN;                 // A@A result (f32)
  __bf16* Abf   = (__bf16*)(Bbuf + NN);      // bf16 row-major copy (A-frag side)
  u32*    Apk   = (u32*)(Abf + NN);          // bf16 K-pair packed copy (B-frag side)
  float*  Xa    = (float*)(Apk + NN / 2);    // features (ping)
  float*  Xb    = Xa + (size_t)N * CH;       // features (pong)
  float*  Ybuf  = Xb + (size_t)N * CH;       // GEMM temp
  float*  Zbuf  = Ybuf + (size_t)N * CH;     // GEMM temp
  float*  dinv  = Zbuf + (size_t)N * CH;     // N
  float*  score = dinv + N;                  // N
  float*  vals  = score + N;                 // N
  int*    perm  = (int*)(vals + N);          // N
  float*  pnorm = (float*)(perm + N);        // 1 (+pad)
  float*  emb   = pnorm + 4;                 // CH

  const int T = 256;

  // build dense adjacency
  k_zero<<<(unsigned)((NN + T - 1) / T), T, 0, stream>>>(Abuf, NN);
  k_scatter_edges<<<(E + T - 1) / T, T, 0, stream>>>(ei, E, Abuf, N);

  // GCN level 0
  launch_gcn(stream, Abuf, N, xin, w0, b0, Xa, Ybuf, Zbuf, dinv);

  const float* Ws[2] = {w1, w2};
  const float* Bv[2] = {b1, b2};
  const float* Pv[2] = {p1, p2};

  int n = N;
  for (int l = 0; l < 2; ++l) {
    const int k = n / 2;
    // augment: diag=1 (during bf16 convert+pack), A@A (bf16 WMMA, exact for
    // small-integer entries), then diag=0
    {
      size_t tot = (size_t)(n / 2) * n;
      k_pack_bf16_diag1<<<(unsigned)((tot + T - 1) / T), T, 0, stream>>>(Abuf, Abf, Apk, n);
    }
    gemm_bf16_wmma<<<dim3(n / 32, n / 32), 128, 0, stream>>>(Abf, Apk, Bbuf, n, n, n, n);
    k_diag_set<<<(n + T - 1) / T, T, 0, stream>>>(Bbuf, n, n, 0.0f);
    // top-k pooling
    k_vec_norm<<<1, 256, 0, stream>>>(Pv[l], pnorm);
    k_row_score<<<n, 256, 0, stream>>>(Xa, Pv[l], pnorm, score, CH);
    k_topk_rank<<<(n + T - 1) / T, T, 0, stream>>>(score, n, k, perm, vals);
    k_gather_x<<<k, 256, 0, stream>>>(Xa, perm, vals, Xb, CH);
    k_gather_adj<<<dim3(k / 256, k), 256, 0, stream>>>(Bbuf, perm, n, Abuf, k);
    // GCN level l+1 (reads Xb, writes Xa)
    launch_gcn(stream, Abuf, k, Xb, Ws[l], Bv[l], Xa, Ybuf, Zbuf, dinv);
    n = k;
  }

  // readout: mean -> L2 normalize -> logits -> log_softmax
  k_col_mean<<<CH, 256, 0, stream>>>(Xa, emb, n, CH);
  k_finalize<<<1, 256, 0, stream>>>(emb, wc, bc, (float*)d_out, CH, NCLS);
}
